// MultiheadAttention_43508018708882
// MI455X (gfx1250) — matrix operand order repaired and checked by reading, not compiled, not run on hardware
//
#include <hip/hip_runtime.h>

// ---------------------------------------------------------------------------
// MI455X (gfx1250) multi-head attention forward.
// All matmuls on v_wmma_f32_16x16x32_bf16 (f32 accumulate). V is materialized
// pre-transposed per head so flash attention is barrier-free; output
// projection stages its A tile with the Tensor Data Mover when available.
// ---------------------------------------------------------------------------

typedef __attribute__((ext_vector_type(16))) __bf16       v16bf;
typedef __attribute__((ext_vector_type(8)))  float        v8f;
typedef __attribute__((ext_vector_type(4)))  unsigned int u32x4;
typedef __attribute__((ext_vector_type(4)))  float        f32x4;

// TDM descriptor vector types (probe-verified builtin signature).
typedef unsigned int tdm_u32x4 __attribute__((ext_vector_type(4)));
typedef int          tdm_i32x4 __attribute__((ext_vector_type(4)));
typedef int          tdm_i32x8 __attribute__((ext_vector_type(8)));

#if __has_builtin(__builtin_amdgcn_tensor_load_to_lds) && __has_builtin(__builtin_amdgcn_s_wait_tensorcnt)
#define HAVE_TDM 1
#endif

#define EMB   1024
#define MROWS 4096   // B*S
#define SEQ   2048
#define HEADS 16
#define HD    64
#define LPAD  40     // LDS row pitch (elements): 80B -> conflict-free 16-lane strides

union FragU { u32x4 q[2]; v16bf v; };
union Pack8 { u32x4 q;    __bf16 h[8]; };

// A-fragment (16x32 bf16, MxK): lane row = lane&15, K-group = lane>>4.
// Elements are two contiguous 8-element chunks at k=kg*8 and k=16+kg*8.
__device__ __forceinline__ v16bf load_a_frag(const __bf16* base, int stride, int lane) {
  const int rowA = lane & 15;
  const int kg   = (lane >> 4) & 1;
  const __bf16* p = base + rowA * stride + kg * 8;
  FragU f;
  f.q[0] = *(const u32x4*)(p);
  f.q[1] = *(const u32x4*)(p + 16);
  return f.v;
}

// B-fragment (32x16 bf16, KxN): lane col = lane&15, K-group = lane>>4.
// Elements are two contiguous 8-element chunks at k=kg*8 and k=16+kg*8.
__device__ __forceinline__ v16bf load_b_frag(const __bf16* base, int stride, int lane) {
  const int col = lane & 15;
  const int ko  = (lane >> 4) << 3;
  const __bf16* p = base + col * stride + ko;
  FragU f;
  f.q[0] = *(const u32x4*)(p);
  f.q[1] = *(const u32x4*)(p + 16);
  return f.v;
}

__device__ __forceinline__ v8f wmma_bf16(v16bf a, v16bf b, v8f c) {
  return __builtin_amdgcn_wmma_f32_16x16x32_bf16(
      /*neg_a=*/false, a, /*neg_b=*/false, b,
      /*c_mod=*/(short)0, c, /*reuse_a=*/false, /*reuse_b=*/false);
}

// ---------------------------------------------------------------------------
// Kernel 1: fused QKV projection. Y_bf16 = X_f32 @ W_f32^T + b  (per blockIdx.z)
// Q,K stored [B*S, EMB] row-major; V stored transposed per head: Vt[b,h,d,s].
// Block tile: 64(M) x 128(N), BK=32. 256 threads = 8 waves (2x4 wave grid).
// ---------------------------------------------------------------------------
__global__ __launch_bounds__(256) void mha_proj_qkv(
    const float* __restrict__ q,  const float* __restrict__ k,  const float* __restrict__ v,
    const float* __restrict__ Wq, const float* __restrict__ Wk, const float* __restrict__ Wv,
    const float* __restrict__ bq, const float* __restrict__ bk, const float* __restrict__ bv,
    __bf16* __restrict__ Qh, __bf16* __restrict__ Kh, __bf16* __restrict__ Vt) {

  const int z = blockIdx.z;
  const float* X    = (z == 0) ? q  : (z == 1) ? k  : v;
  const float* W    = (z == 0) ? Wq : (z == 1) ? Wk : Wv;
  const float* bias = (z == 0) ? bq : (z == 1) ? bk : bv;
  __bf16*      Y    = (z == 0) ? Qh : (z == 1) ? Kh : Vt;

  __shared__ __attribute__((aligned(16))) __bf16 As[64][LPAD];
  __shared__ __attribute__((aligned(16))) __bf16 Bs[128][LPAD];

  const int tid  = threadIdx.x;
  const int lane = tid & 31;
  const int wave = tid >> 5;       // 0..7
  const int wm   = wave & 1;       // 2 M-groups of 32
  const int wn   = wave >> 1;      // 4 N-groups of 32
  const int m0   = blockIdx.y * 64;
  const int n0   = blockIdx.x * 128;

  const v8f vzero = {0.f,0.f,0.f,0.f,0.f,0.f,0.f,0.f};
  v8f acc[2][2] = {{vzero, vzero}, {vzero, vzero}};

  const int arow  = tid >> 2;        // 0..63
  const int akseg = (tid & 3) * 8;   // 0,8,16,24
  const int brow  = tid >> 1;        // 0..127
  const int bkseg = (tid & 1) * 16;  // 0,16

  for (int k0 = 0; k0 < EMB; k0 += 32) {
    // Stage A (64x32 f32 -> bf16), 8 floats/thread, coalesced b128 loads.
    {
      const float* pa = X + (m0 + arow) * EMB + k0 + akseg;
      f32x4 a0 = *(const f32x4*)(pa);
      f32x4 a1 = *(const f32x4*)(pa + 4);
      if (k0 + 32 < EMB) __builtin_prefetch(pa + 32, 0, 3);
      Pack8 pk;
      pk.h[0] = (__bf16)a0.x; pk.h[1] = (__bf16)a0.y; pk.h[2] = (__bf16)a0.z; pk.h[3] = (__bf16)a0.w;
      pk.h[4] = (__bf16)a1.x; pk.h[5] = (__bf16)a1.y; pk.h[6] = (__bf16)a1.z; pk.h[7] = (__bf16)a1.w;
      *(u32x4*)&As[arow][akseg] = pk.q;
    }
    // Stage B (128x32 f32 -> bf16), 16 floats/thread.
    {
      const float* pb = W + (n0 + brow) * EMB + k0 + bkseg;
      f32x4 b0 = *(const f32x4*)(pb);
      f32x4 b1 = *(const f32x4*)(pb + 4);
      f32x4 b2 = *(const f32x4*)(pb + 8);
      f32x4 b3 = *(const f32x4*)(pb + 12);
      if (k0 + 32 < EMB) __builtin_prefetch(pb + 32, 0, 3);
      Pack8 p0, p1;
      p0.h[0] = (__bf16)b0.x; p0.h[1] = (__bf16)b0.y; p0.h[2] = (__bf16)b0.z; p0.h[3] = (__bf16)b0.w;
      p0.h[4] = (__bf16)b1.x; p0.h[5] = (__bf16)b1.y; p0.h[6] = (__bf16)b1.z; p0.h[7] = (__bf16)b1.w;
      p1.h[0] = (__bf16)b2.x; p1.h[1] = (__bf16)b2.y; p1.h[2] = (__bf16)b2.z; p1.h[3] = (__bf16)b2.w;
      p1.h[4] = (__bf16)b3.x; p1.h[5] = (__bf16)b3.y; p1.h[6] = (__bf16)b3.z; p1.h[7] = (__bf16)b3.w;
      *(u32x4*)&Bs[brow][bkseg]     = p0.q;
      *(u32x4*)&Bs[brow][bkseg + 8] = p1.q;
    }
    __syncthreads();

    const __bf16* abase = &As[wm * 32][0];
    const __bf16* bbase = &Bs[wn * 32][0];
    v16bf a0f = load_a_frag(abase,             LPAD, lane);
    v16bf a1f = load_a_frag(abase + 16 * LPAD, LPAD, lane);
    v16bf b0f = load_b_frag(bbase,             LPAD, lane);
    v16bf b1f = load_b_frag(bbase + 16 * LPAD, LPAD, lane);
    acc[0][0] = wmma_bf16(a0f, b0f, acc[0][0]);
    acc[0][1] = wmma_bf16(a0f, b1f, acc[0][1]);
    acc[1][0] = wmma_bf16(a1f, b0f, acc[1][0]);
    acc[1][1] = wmma_bf16(a1f, b1f, acc[1][1]);
    __syncthreads();
  }

  // C/D layout: row = r + 8*(lane>>4), col = lane&15. Bias fused, bf16 out.
  const int hi = lane >> 4;
  const int ln = lane & 15;
#pragma unroll
  for (int mi = 0; mi < 2; ++mi)
#pragma unroll
    for (int ni = 0; ni < 2; ++ni) {
      const int colg = n0 + wn * 32 + ni * 16 + ln;
      const float bcol = bias[colg];
      const int row0 = m0 + wm * 32 + mi * 16 + 8 * hi;   // 8 consecutive rows
      if (z != 2) {
#pragma unroll
        for (int r = 0; r < 8; ++r)
          Y[(row0 + r) * EMB + colg] = (__bf16)(acc[mi][ni][r] + bcol);
      } else {
        // V: store transposed per head -> Vt[((b*H + h)*64 + d) * SEQ + s],
        // 8 consecutive s per lane => one b128 store.
        Pack8 pk;
#pragma unroll
        for (int r = 0; r < 8; ++r) pk.h[r] = (__bf16)(acc[mi][ni][r] + bcol);
        const size_t off =
            ((size_t)((row0 >> 11) * HEADS + (colg >> 6)) * HD + (colg & 63)) * SEQ
            + (row0 & (SEQ - 1));
        *(u32x4*)(Y + off) = pk.q;
      }
    }
}

// ---------------------------------------------------------------------------
// Kernel 2: causal flash attention, barrier-free.
// grid = (S/64, HEADS, B), block = 128 (4 waves); wave w owns 16 query rows
// and loops only over its own causal key-tile range. K fragments and
// (pre-transposed) V fragments load straight from global as b128s; the P
// C->A layout fix-up goes through a private per-wave LDS tile (in-order DS).
// ---------------------------------------------------------------------------
__global__ __launch_bounds__(128) void mha_flash_attn(
    const __bf16* __restrict__ Qh, const __bf16* __restrict__ Kh,
    const __bf16* __restrict__ Vt, __bf16* __restrict__ O) {

  __shared__ __attribute__((aligned(16))) __bf16 ps[4][16][LPAD];  // per-wave P tile

  const int tid   = threadIdx.x;
  const int lane  = tid & 31;
  const int wave  = tid >> 5;            // 0..3
  const int qtile = blockIdx.x;          // tile of 64 query rows
  const int h     = blockIdx.y;
  const int b     = blockIdx.z;

  const int base  = (b * SEQ) * EMB + h * HD;
  const int qrow0 = qtile * 64 + wave * 16;
  const int hi    = lane >> 4;
  const int ln    = lane & 15;

  // Q fragments, loaded once (direct from global; b128-friendly layout).
  const __bf16* qbase = Qh + base + qrow0 * EMB;
  const v16bf aq0 = load_a_frag(qbase,      EMB, lane);   // k = 0..31
  const v16bf aq1 = load_a_frag(qbase + 32, EMB, lane);   // k = 32..63

  const __bf16* vhead = Vt + (size_t)(b * HEADS + h) * HD * SEQ;  // [d][s], s-contig

  const v8f vzero = {0.f,0.f,0.f,0.f,0.f,0.f,0.f,0.f};
  v8f oacc[4] = {vzero, vzero, vzero, vzero};
  float mrow[8], lrow[8];
#pragma unroll
  for (int r = 0; r < 8; ++r) { mrow[r] = -3.0e38f; lrow[r] = 0.f; }

  const float scale = 0.125f;                      // 1/sqrt(64)
  const float L2E   = 1.44269504088896340736f;
  const int   njt   = ((qrow0 + 15) >> 5) + 1;     // per-wave causal tile count

  for (int jt = 0; jt < njt; ++jt) {
    const int kbase = jt * 32;

    // ---- S = Q @ K^T for this 16x32 tile (K fragments direct from global) ----
    const __bf16* kb0 = Kh + base + kbase * EMB;
    const __bf16* kb1 = kb0 + 16 * EMB;
    v16bf bk00 = load_b_frag(kb0,      EMB, lane);
    v16bf bk01 = load_b_frag(kb0 + 32, EMB, lane);
    v16bf bk10 = load_b_frag(kb1,      EMB, lane);
    v16bf bk11 = load_b_frag(kb1 + 32, EMB, lane);
    if (jt + 1 < njt) {
      __builtin_prefetch(kb0 + 32 * EMB, 0, 3);            // next K tile
      __builtin_prefetch(vhead + ln * SEQ + kbase + 32, 0, 3);  // next V tile
    }
    v8f s0 = vzero, s1 = vzero;
    s0 = wmma_bf16(aq0, bk00, s0); s0 = wmma_bf16(aq1, bk01, s0);
    s1 = wmma_bf16(aq0, bk10, s1); s1 = wmma_bf16(aq1, bk11, s1);

    // ---- online softmax (row reductions across the 16-lane half-groups) ----
#pragma unroll
    for (int r = 0; r < 8; ++r) {
      const int qi = qrow0 + 8 * hi + r;
      const int c0 = kbase + ln;
      const int c1 = kbase + 16 + ln;
      float x0 = (c0 <= qi) ? s0[r] * scale : -3.0e38f;
      float x1 = (c1 <= qi) ? s1[r] * scale : -3.0e38f;
      float mx = fmaxf(x0, x1);
#pragma unroll
      for (int off = 1; off < 16; off <<= 1)
        mx = fmaxf(mx, __shfl_xor(mx, off, 32));
      const float newm = fmaxf(mrow[r], mx);
      const float corr = exp2f((mrow[r] - newm) * L2E);
      const float p0 = exp2f((x0 - newm) * L2E);
      const float p1 = exp2f((x1 - newm) * L2E);
      float ss = p0 + p1;
#pragma unroll
      for (int off = 1; off < 16; off <<= 1)
        ss += __shfl_xor(ss, off, 32);
      lrow[r] = lrow[r] * corr + ss;
      mrow[r] = newm;
#pragma unroll
      for (int t = 0; t < 4; ++t) oacc[t][r] *= corr;
      const int prow = 8 * hi + r;
      ps[wave][prow][ln]      = (__bf16)p0;    // C-layout -> A-layout via LDS
      ps[wave][prow][16 + ln] = (__bf16)p1;
    }

    // ---- O += P @ V (V^T fragments straight from global) ----
    const v16bf ap = load_a_frag(&ps[wave][0][0], LPAD, lane);
#pragma unroll
    for (int t = 0; t < 4; ++t) {
      v16bf bv = load_b_frag(vhead + (size_t)(t * 16) * SEQ + kbase, SEQ, lane);
      oacc[t] = wmma_bf16(ap, bv, oacc[t]);
    }
  }

  // Normalize and store O (bf16, [B*S, EMB] with head offset).
#pragma unroll
  for (int r = 0; r < 8; ++r) {
    const float inv = 1.0f / lrow[r];
    const int rowg = qrow0 + 8 * hi + r;
    __bf16* po = O + base + rowg * EMB;
#pragma unroll
    for (int t = 0; t < 4; ++t)
      po[t * 16 + ln] = (__bf16)(oacc[t][r] * inv);
  }
}

// ---------------------------------------------------------------------------
// Kernel 3: output projection. out_f32 = O_bf16 @ Wo_f32^T + bo.
// A tile (pure bf16 copy, padded pitch) staged by the Tensor Data Mover when
// the builtin is available: 2D tile 64x32, stride 1024, pad 4 DWORDs every
// 16 DWORDs -> exactly the As[64][40] layout. One wave issues, waits
// s_wait_tensorcnt 0, block barrier publishes.
// ---------------------------------------------------------------------------
__global__ __launch_bounds__(256) void mha_out_proj(
    const __bf16* __restrict__ A, const float* __restrict__ W,
    const float* __restrict__ bias, float* __restrict__ Y) {

  __shared__ __attribute__((aligned(16))) __bf16 As[64][LPAD];
  __shared__ __attribute__((aligned(16))) __bf16 Bs[128][LPAD];

  const int tid  = threadIdx.x;
  const int lane = tid & 31;
  const int wave = tid >> 5;
  const int wm   = wave & 1;
  const int wn   = wave >> 1;
  const int m0   = blockIdx.y * 64;
  const int n0   = blockIdx.x * 128;

  const v8f vzero = {0.f,0.f,0.f,0.f,0.f,0.f,0.f,0.f};
  v8f acc[2][2] = {{vzero, vzero}, {vzero, vzero}};

  const int arow  = tid >> 2;
  const int akseg = (tid & 3) * 8;
  const int brow  = tid >> 1;
  const int bkseg = (tid & 1) * 16;
  (void)arow; (void)akseg;

  for (int k0 = 0; k0 < EMB; k0 += 32) {
#if defined(HAVE_TDM)
    if (wave == 0) {
      // D# group0: count=1, lds_addr, 57-bit global_addr, type=2 ("image").
      const unsigned lds_off =
          (unsigned)(unsigned long long)(__attribute__((address_space(3))) __bf16*)&As[0][0];
      const unsigned long long ga =
          (unsigned long long)(A + (size_t)m0 * EMB + k0);
      tdm_u32x4 g0 = { 1u, lds_off, (unsigned)(ga & 0xffffffffu),
                       (unsigned)(((ga >> 32) & 0x01ffffffu) | (2u << 30)) };
      // D# group1: data_size=2B, pad_enable, pad_interval=16 DW, pad_amount=4 DW,
      // tensor_dim0=32, tensor_dim1=64, tile_dim0=32, tile_dim1=64,
      // tensor_dim0_stride=1024.
      tdm_i32x8 g1 = { (int)((1u << 16) | (1u << 20) | (3u << 22) | (3u << 25)),
                       (int)(32u << 16),   // tensor_dim0 low16 @ [63:48]
                       (int)(64u << 16),   // tensor_dim1 low16 @ [95:80]
                       (int)(32u << 16),   // tile_dim0 @ [127:112]
                       64,                 // tile_dim1 @ [143:128]
                       1024,               // tensor_dim0_stride low32 @ [191:160]
                       0, 0 };
      tdm_i32x4 gz = {0, 0, 0, 0};
#if __has_include(<hip/amd_detail/amd_gfx1250_TDM.h>)
      tdm_i32x8 gz8 = {0, 0, 0, 0, 0, 0, 0, 0};
      __builtin_amdgcn_tensor_load_to_lds(g0, g1, gz, gz, gz8, 0);
#else
      __builtin_amdgcn_tensor_load_to_lds(g0, g1, gz, gz, 0);
#endif
    }
#else
    // Fallback: manual bf16 copy, 16B per thread.
    *(u32x4*)&As[arow][akseg] = *(const u32x4*)(A + (m0 + arow) * EMB + k0 + akseg);
#endif
    // Wo fp32 -> bf16 (all waves).
    {
      const float* pb = W + (n0 + brow) * EMB + k0 + bkseg;
      f32x4 b0 = *(const f32x4*)(pb);
      f32x4 b1 = *(const f32x4*)(pb + 4);
      f32x4 b2 = *(const f32x4*)(pb + 8);
      f32x4 b3 = *(const f32x4*)(pb + 12);
      if (k0 + 32 < EMB) __builtin_prefetch(pb + 32, 0, 3);
      Pack8 p0, p1;
      p0.h[0] = (__bf16)b0.x; p0.h[1] = (__bf16)b0.y; p0.h[2] = (__bf16)b0.z; p0.h[3] = (__bf16)b0.w;
      p0.h[4] = (__bf16)b1.x; p0.h[5] = (__bf16)b1.y; p0.h[6] = (__bf16)b1.z; p0.h[7] = (__bf16)b1.w;
      p1.h[0] = (__bf16)b2.x; p1.h[1] = (__bf16)b2.y; p1.h[2] = (__bf16)b2.z; p1.h[3] = (__bf16)b2.w;
      p1.h[4] = (__bf16)b3.x; p1.h[5] = (__bf16)b3.y; p1.h[6] = (__bf16)b3.z; p1.h[7] = (__bf16)b3.w;
      *(u32x4*)&Bs[brow][bkseg]     = p0.q;
      *(u32x4*)&Bs[brow][bkseg + 8] = p1.q;
    }
#if defined(HAVE_TDM)
    if (wave == 0) __builtin_amdgcn_s_wait_tensorcnt(0);
#endif
    __syncthreads();

    const __bf16* abase = &As[wm * 32][0];
    const __bf16* bbase = &Bs[wn * 32][0];
    v16bf a0f = load_a_frag(abase,             LPAD, lane);
    v16bf a1f = load_a_frag(abase + 16 * LPAD, LPAD, lane);
    v16bf b0f = load_b_frag(bbase,             LPAD, lane);
    v16bf b1f = load_b_frag(bbase + 16 * LPAD, LPAD, lane);
    acc[0][0] = wmma_bf16(a0f, b0f, acc[0][0]);
    acc[0][1] = wmma_bf16(a0f, b1f, acc[0][1]);
    acc[1][0] = wmma_bf16(a1f, b0f, acc[1][0]);
    acc[1][1] = wmma_bf16(a1f, b1f, acc[1][1]);
    __syncthreads();
  }

  const int hi = lane >> 4;
  const int ln = lane & 15;
#pragma unroll
  for (int mi = 0; mi < 2; ++mi)
#pragma unroll
    for (int ni = 0; ni < 2; ++ni) {
      const int colg = n0 + wn * 32 + ni * 16 + ln;
      const float bcol = bias[colg];
#pragma unroll
      for (int r = 0; r < 8; ++r) {
        const int rowg = m0 + wm * 32 + mi * 16 + 8 * hi + r;
        Y[rowg * EMB + colg] = acc[mi][ni][r] + bcol;
      }
    }
}

// ---------------------------------------------------------------------------
// Host-side launcher. Inputs (setup_inputs order):
// 0:q 1:k 2:v 3:Wq 4:bq 5:Wk 6:bk 7:Wv 8:bv 9:Wo 10:bo 11:attn_mask (causal; ignored)
// Workspace: Qh | Kh | Vt | O, each 4096*1024 bf16 (32 MB total).
// ---------------------------------------------------------------------------
extern "C" void kernel_launch(void* const* d_in, const int* in_sizes, int n_in,
                              void* d_out, int out_size, void* d_ws, size_t ws_size,
                              hipStream_t stream) {
  (void)in_sizes; (void)n_in; (void)out_size; (void)ws_size;

  const float* q  = (const float*)d_in[0];
  const float* k  = (const float*)d_in[1];
  const float* v  = (const float*)d_in[2];
  const float* Wq = (const float*)d_in[3];
  const float* bq = (const float*)d_in[4];
  const float* Wk = (const float*)d_in[5];
  const float* bk = (const float*)d_in[6];
  const float* Wv = (const float*)d_in[7];
  const float* bv = (const float*)d_in[8];
  const float* Wo = (const float*)d_in[9];
  const float* bo = (const float*)d_in[10];

  const size_t plane = (size_t)MROWS * EMB;   // 4096*1024 elements
  __bf16* Qh = (__bf16*)d_ws;
  __bf16* Kh = Qh + plane;
  __bf16* Vt = Kh + plane;                    // [B, H, D, S] transposed V
  __bf16* Oc = Vt + plane;

  dim3 gProj(EMB / 128, MROWS / 64, 3);       // (8, 64, 3)
  mha_proj_qkv<<<gProj, 256, 0, stream>>>(q, k, v, Wq, Wk, Wv, bq, bk, bv, Qh, Kh, Vt);

  dim3 gAttn(SEQ / 64, HEADS, 2);             // (32, 16, 2)
  mha_flash_attn<<<gAttn, 128, 0, stream>>>(Qh, Kh, Vt, Oc);

  dim3 gOut(EMB / 128, MROWS / 64, 1);        // (8, 64)
  mha_out_proj<<<gOut, 256, 0, stream>>>(Oc, Wo, bo, (float*)d_out);
}
